// MambaDecoder_32968168964659
// MI455X (gfx1250) — matrix operand order repaired and checked by reading, not compiled, not run on hardware
//
#include <hip/hip_runtime.h>

typedef __attribute__((ext_vector_type(16))) __bf16 v16bf;
typedef __attribute__((ext_vector_type(8)))  float  v8f;

#define L_SEQ 1024
#define DMODEL 1024
#define DI 2048
#define NST 16
#define DTR 64
#define XPJ 96      // DTR + 2*N
#define XPJP 128    // XPJ padded to 128
#define E2 4096     // 2*DI

// low 32 bits of a generic LDS pointer == LDS byte offset (ISA 10.2)
__device__ __forceinline__ unsigned lds_addr32(const void* p) {
    return (unsigned)(unsigned long long)p;
}

// ---------------- utility ----------------
__global__ void copy_f32x4_kernel(const float4* __restrict__ src, float4* __restrict__ dst, int n4) {
    int i = blockIdx.x * 256 + threadIdx.x;
    if (i < n4) dst[i] = src[i];
}

// ---------------- weight convert+transpose: Bt[N,K] f32 -> Bc[K,Npad] bf16 (zero-padded) ----
__global__ __launch_bounds__(256) void wconv_kernel(const float* __restrict__ Bt,
                                                    __bf16* __restrict__ Bc,
                                                    int N, int K, int Npad) {
    __shared__ float t[32][33];
    const int k0 = blockIdx.x * 32;
    const int n0 = blockIdx.y * 32;
    const int tx = threadIdx.x;        // 32
    const int ty = threadIdx.y;        // 8
#pragma unroll
    for (int j = 0; j < 4; ++j) {
        const int n = n0 + ty + 8 * j;
        t[ty + 8 * j][tx] = (n < N) ? Bt[(size_t)n * K + (k0 + tx)] : 0.f;
    }
    __syncthreads();
#pragma unroll
    for (int j = 0; j < 4; ++j) {
        const int k = ty + 8 * j;
        Bc[(size_t)(k0 + k) * Npad + (n0 + tx)] = (__bf16)t[tx][k];
    }
}

// ---------------- RMSNorm ----------------
__global__ __launch_bounds__(256) void rmsnorm_kernel(const float* __restrict__ x,
                                                      const float* __restrict__ w,
                                                      float* __restrict__ xn) {
    __shared__ float red[256];
    const int row = blockIdx.x;
    const float4* xr = reinterpret_cast<const float4*>(x + (size_t)row * DMODEL);
    float4 v = xr[threadIdx.x];                       // 256 threads x 4 = 1024
    float s = v.x * v.x + v.y * v.y + v.z * v.z + v.w * v.w;
    red[threadIdx.x] = s; __syncthreads();
    for (int off = 128; off > 0; off >>= 1) {
        if (threadIdx.x < off) red[threadIdx.x] += red[threadIdx.x + off];
        __syncthreads();
    }
    const float rs = rsqrtf(red[0] * (1.f / DMODEL) + 1e-5f);
    const float4* wr = reinterpret_cast<const float4*>(w);
    float4 wv = wr[threadIdx.x];
    float4 o; o.x = v.x * rs * wv.x; o.y = v.y * rs * wv.y;
    o.z = v.z * rs * wv.z; o.w = v.w * rs * wv.w;
    reinterpret_cast<float4*>(xn + (size_t)row * DMODEL)[threadIdx.x] = o;
}

// ---------------- WMMA GEMM: C[M,N] (+=) act(A[M,K] * B[K,N] + bias) ----------------
// A: f32 row-major stride lda. Bc: bf16 column-padded [K, ldb] (pre-transposed weights).
// Block tile 64(M) x 128(N), BK=32, double-buffered LDS.
// B tile: GLOBAL_LOAD_ASYNC_TO_LDS_B128 (memory->LDS, no VGPR transit, ASYNCcnt).
// A tile: register path with f32->bf16 packed converts.
// 8 waves as 2(M) x 4(N); each wave owns a 32x32 slab (4 WMMAs / K-step).
// GUARD_N: epilogue store guard (only needed when N % 128 != 0; loads are pad-safe).
template <bool GUARD_N>
__global__ __launch_bounds__(256) void gemm_wmma_bf16(const float* __restrict__ A, int lda,
                                                      const __bf16* __restrict__ Bc, int ldb,
                                                      float* __restrict__ C,
                                                      const float* __restrict__ bias,
                                                      int M, int N, int K, int act, int accum) {
    __shared__ alignas(16) __bf16 As[2][64][40];    // 64 x 32 (+pad): row = 80B
    __shared__ alignas(16) __bf16 Bs[2][32][152];   // 32 x 128 (+pad): row = 304B

    const int tid  = threadIdx.x;
    const int lane = tid & 31;
    const int wave = tid >> 5;
    const int wm   = wave & 1;        // 2 waves along M (32 each)
    const int wn   = wave >> 1;       // 4 waves along N (32 each)
    const int mBlk = blockIdx.y * 64;
    const int nBlk = blockIdx.x * 128;

    // per-thread staging coordinates (fixed across K-steps)
    const int ar[2]  = { (tid + 0)   >> 3, (tid + 256) >> 3 };   // A: row, float4-col
    const int ac4[2] = { (tid + 0)   & 7,  (tid + 256) & 7  };
    const int bk[2]  = { (tid + 0)   >> 4, (tid + 256) >> 4 };   // B: k-row, 8-elt chunk
    const int bc8[2] = { (tid + 0)   & 15, (tid + 256) & 15 };

    float4 ra[2];

    auto asyncLoadB = [&](int k0, int buf) {            // global -> LDS, no VGPR transit
#pragma unroll
        for (int i = 0; i < 2; ++i) {
            const unsigned long long ga = (unsigned long long)(
                Bc + (size_t)(k0 + bk[i]) * ldb + (nBlk + 8 * bc8[i]));
            const unsigned la = lds_addr32(&Bs[buf][bk[i]][8 * bc8[i]]);
            asm volatile("global_load_async_to_lds_b128 %0, %1, off"
                         :: "v"(la), "v"(ga) : "memory");
        }
    };
    auto loadA = [&](int k0) {
#pragma unroll
        for (int i = 0; i < 2; ++i)
            ra[i] = *reinterpret_cast<const float4*>(
                A + (size_t)(mBlk + ar[i]) * lda + (k0 + 4 * ac4[i]));
    };
    auto storeA = [&](int buf) {
#pragma unroll
        for (int i = 0; i < 2; ++i) {
            union { __bf16 h[4]; unsigned long long u; } p;
            p.h[0] = (__bf16)ra[i].x; p.h[1] = (__bf16)ra[i].y;
            p.h[2] = (__bf16)ra[i].z; p.h[3] = (__bf16)ra[i].w;
            *reinterpret_cast<unsigned long long*>(&As[buf][ar[i]][4 * ac4[i]]) = p.u;
        }
    };

    v8f acc00 = {}, acc01 = {}, acc10 = {}, acc11 = {};

    const int nTiles = K >> 5;
    asyncLoadB(0, 0);
    loadA(0);
    storeA(0);
    asm volatile("s_wait_asynccnt 0x0" ::: "memory");
    __syncthreads();

    const int mrow = 32 * wm + (lane & 15);
    const int kb   = (lane >> 4) << 3;
    const int nOff = 32 * wn;

    for (int t = 0; t < nTiles; ++t) {
        const int cur = t & 1;
        const bool more = (t + 1) < nTiles;
        if (more) {
            asyncLoadB(32 * (t + 1), cur ^ 1);   // async copy overlaps WMMAs
            loadA(32 * (t + 1));                 // register prefetch overlaps WMMAs
        }

        v16bf af0, af1;
#pragma unroll
        for (int i = 0; i < 8; ++i) {
            af0[i]     = As[cur][mrow][kb + i];      af0[i + 8] = As[cur][mrow][kb + 16 + i];
            af1[i]     = As[cur][mrow + 16][kb + i]; af1[i + 8] = As[cur][mrow + 16][kb + 16 + i];
        }
        v16bf b0, b1;
#pragma unroll
        for (int j = 0; j < 16; ++j) {
            b0[j] = Bs[cur][kb + j + ((j >> 3) << 3)][nOff + (lane & 15)];
            b1[j] = Bs[cur][kb + j + ((j >> 3) << 3)][nOff + 16 + (lane & 15)];
        }
        acc00 = __builtin_amdgcn_wmma_f32_16x16x32_bf16(false, af0, false, b0, (short)0, acc00, false, false);
        acc01 = __builtin_amdgcn_wmma_f32_16x16x32_bf16(false, af0, false, b1, (short)0, acc01, false, false);
        acc10 = __builtin_amdgcn_wmma_f32_16x16x32_bf16(false, af1, false, b0, (short)0, acc10, false, false);
        acc11 = __builtin_amdgcn_wmma_f32_16x16x32_bf16(false, af1, false, b1, (short)0, acc11, false, false);

        if (more) {
            storeA(cur ^ 1);
            asm volatile("s_wait_asynccnt 0x0" ::: "memory");   // drain B copy before barrier
        }
        __syncthreads();
    }

    // ---- epilogue: D layout m = base + r + 8*(lane>=16), n = lane&15 ----
    const int mBase = mBlk + 32 * wm + ((lane >> 4) << 3);
    const int nBase = nBlk + 32 * wn + (lane & 15);
#pragma unroll
    for (int mi = 0; mi < 2; ++mi) {
#pragma unroll
        for (int ni = 0; ni < 2; ++ni) {
            const v8f& acc = mi == 0 ? (ni == 0 ? acc00 : acc01) : (ni == 0 ? acc10 : acc11);
            const int n = nBase + 16 * ni;
            if (!GUARD_N || n < N) {
                const float bv = bias ? bias[n] : 0.f;
#pragma unroll
                for (int r = 0; r < 8; ++r) {
                    const int m = mBase + 16 * mi + r;
                    float v = acc[r] + bv;
                    if (act == 1) v = (v > 20.f) ? v : log1pf(__expf(v));
                    size_t idx = (size_t)m * N + n;
                    C[idx] = accum ? (C[idx] + v) : v;
                }
            }
        }
    }
}

// ---------------- causal depthwise conv (K=4) + bias + SiLU ----------------
__global__ __launch_bounds__(256) void conv_silu_kernel(const float* __restrict__ xz,
                                                        const float* __restrict__ w,
                                                        const float* __restrict__ b,
                                                        float* __restrict__ u) {
    const int idx = blockIdx.x * 256 + threadIdx.x;   // L*DI threads
    const int l = idx >> 11;          // / DI
    const int d = idx & (DI - 1);
    float acc = b[d];
#pragma unroll
    for (int k = 0; k < 4; ++k) {
        const int ls = l - 3 + k;
        if (ls >= 0) acc += xz[(size_t)ls * E2 + d] * w[d * 4 + k];
    }
    const float sig = 1.f / (1.f + __expf(-acc));
    u[idx] = acc * sig;
}

// ---------------- fused selective scan + D-skip + SiLU(z) gate ----------------
// One lane per channel d (2048 lanes). Coalesced [L,DI] loads; B/C broadcast from x_dbl.
__global__ __launch_bounds__(256) void scan_kernel(const float* __restrict__ delta,
                                                   const float* __restrict__ u,
                                                   const float* __restrict__ xz,
                                                   const float* __restrict__ xdbl,
                                                   const float* __restrict__ A_log_l,
                                                   const float* __restrict__ D_skip_l,
                                                   float* __restrict__ y) {
    const int d = blockIdx.x * 256 + threadIdx.x;     // 0..DI-1
    float a[NST], h[NST];
#pragma unroll
    for (int n = 0; n < NST; ++n) {
        a[n] = -__expf(A_log_l[(size_t)d * NST + n]);
        h[n] = 0.f;
    }
    const float dsk = D_skip_l[d];
    for (int l = 0; l < L_SEQ; ++l) {
        const float del = delta[(size_t)l * DI + d];
        const float uv  = u[(size_t)l * DI + d];
        const float zv  = xz[(size_t)l * E2 + DI + d];
        const float dbu = del * uv;
        // broadcast B[l][:], C[l][:] as float4 (uniform addresses)
        const float4* bc4 = reinterpret_cast<const float4*>(xdbl + (size_t)l * XPJ + DTR);
        float bb[NST], cc[NST];
#pragma unroll
        for (int q = 0; q < 4; ++q) {
            float4 bv = bc4[q], cv = bc4[4 + q];
            bb[4 * q + 0] = bv.x; bb[4 * q + 1] = bv.y; bb[4 * q + 2] = bv.z; bb[4 * q + 3] = bv.w;
            cc[4 * q + 0] = cv.x; cc[4 * q + 1] = cv.y; cc[4 * q + 2] = cv.z; cc[4 * q + 3] = cv.w;
        }
        float acc = 0.f;
#pragma unroll
        for (int n = 0; n < NST; ++n) {
            const float dA = __expf(del * a[n]);
            h[n] = dA * h[n] + dbu * bb[n];
            acc += h[n] * cc[n];
        }
        acc += uv * dsk;
        const float sig = 1.f / (1.f + __expf(-zv));
        y[(size_t)l * DI + d] = acc * (zv * sig);
    }
}

// ---------------- host orchestration ----------------
extern "C" void kernel_launch(void* const* d_in, const int* in_sizes, int n_in,
                              void* d_out, int out_size, void* d_ws, size_t ws_size,
                              hipStream_t stream) {
    const float* x_in    = (const float*)d_in[0];
    const float* norm_w  = (const float*)d_in[1];
    const float* in_w    = (const float*)d_in[2];
    const float* conv_w  = (const float*)d_in[3];
    const float* conv_b  = (const float*)d_in[4];
    const float* xproj_w = (const float*)d_in[5];
    const float* dt_w    = (const float*)d_in[6];
    const float* dt_b    = (const float*)d_in[7];
    const float* A_log   = (const float*)d_in[8];
    const float* D_skip  = (const float*)d_in[9];
    const float* out_w   = (const float*)d_in[10];

    float* xcur = (float*)d_out;                  // running residual state [L, D]

    float* ws = (float*)d_ws;
    float* xn    = ws;                            size_t off = (size_t)L_SEQ * DMODEL;
    float* xz    = ws + off;                      off += (size_t)L_SEQ * E2;
    float* u     = ws + off;                      off += (size_t)L_SEQ * DI;
    float* xdbl  = ws + off;                      off += (size_t)L_SEQ * XPJ;
    float* delta = ws + off;                      off += (size_t)L_SEQ * DI;
    float* ybuf  = ws + off;                      off += (size_t)L_SEQ * DI;
    __bf16* wbc  = (__bf16*)(ws + off);           // up to K*Npad bf16 (max 1024*4096)

    // x -> d_out (residual accumulator)
    copy_f32x4_kernel<<<(L_SEQ * DMODEL / 4 + 255) / 256, 256, 0, stream>>>(
        (const float4*)x_in, (float4*)xcur, L_SEQ * DMODEL / 4);

    const dim3 tb(32, 8);

    for (int layer = 0; layer < 4; ++layer) {
        const float* nw  = norm_w  + (size_t)layer * DMODEL;
        const float* iw  = in_w    + (size_t)layer * E2 * DMODEL;
        const float* cw  = conv_w  + (size_t)layer * DI * 4;
        const float* cb  = conv_b  + (size_t)layer * DI;
        const float* xpw = xproj_w + (size_t)layer * XPJ * DI;
        const float* dw  = dt_w    + (size_t)layer * DI * DTR;
        const float* db  = dt_b    + (size_t)layer * DI;
        const float* al  = A_log   + (size_t)layer * DI * NST;
        const float* dsk = D_skip  + (size_t)layer * DI;
        const float* ow  = out_w   + (size_t)layer * DMODEL * DI;

        rmsnorm_kernel<<<L_SEQ, 256, 0, stream>>>(xcur, nw, xn);

        // xz[L, 4096] = xn[L,1024] @ in_w^T
        wconv_kernel<<<dim3(DMODEL / 32, E2 / 32), tb, 0, stream>>>(iw, wbc, E2, DMODEL, E2);
        gemm_wmma_bf16<false><<<dim3(E2 / 128, L_SEQ / 64), 256, 0, stream>>>(
            xn, DMODEL, wbc, E2, xz, nullptr, L_SEQ, E2, DMODEL, 0, 0);

        // u = silu(depthwise_conv(xz[:, :DI]) + b)
        conv_silu_kernel<<<(L_SEQ * DI) / 256, 256, 0, stream>>>(xz, cw, cb, u);

        // x_dbl[L, 96] = u @ xproj_w^T   (single 128-wide block column; padded weights)
        wconv_kernel<<<dim3(DI / 32, XPJP / 32), tb, 0, stream>>>(xpw, wbc, XPJ, DI, XPJP);
        gemm_wmma_bf16<true><<<dim3(1, L_SEQ / 64), 256, 0, stream>>>(
            u, DI, wbc, XPJP, xdbl, nullptr, L_SEQ, XPJ, DI, 0, 0);

        // delta[L, 2048] = softplus(x_dbl[:, :64] @ dt_w^T + dt_b)
        wconv_kernel<<<dim3(DTR / 32, DI / 32), tb, 0, stream>>>(dw, wbc, DI, DTR, DI);
        gemm_wmma_bf16<false><<<dim3(DI / 128, L_SEQ / 64), 256, 0, stream>>>(
            xdbl, XPJ, wbc, DI, delta, db, L_SEQ, DI, DTR, 1, 0);

        // y = (selective_scan + u*D) * silu(z)
        scan_kernel<<<DI / 256, 256, 0, stream>>>(delta, u, xz, xdbl, al, dsk, ybuf);

        // xcur += y @ out_w^T   (residual)
        wconv_kernel<<<dim3(DI / 32, DMODEL / 32), tb, 0, stream>>>(ow, wbc, DMODEL, DI, DMODEL);
        gemm_wmma_bf16<false><<<dim3(DMODEL / 128, L_SEQ / 64), 256, 0, stream>>>(
            ybuf, DI, wbc, DMODEL, xcur, nullptr, L_SEQ, DMODEL, DI, 0, 1);
    }
}
